// CustomMultiHeadAttention_54966991454645
// MI455X (gfx1250) — compile-verified, matrix operand order checked by reading, not run on hardware
//
#include <hip/hip_runtime.h>
#include <hip/hip_bf16.h>
#include <math.h>

// MHA: B=2, S=2048, D=1024, H=16, HD=64.
// bf16 WMMA + fp32 softmax; all tile movement via Tensor Data Mover (TDM),
// double-buffered LDS, pipelined against WMMA compute.

typedef __attribute__((ext_vector_type(16))) __bf16 v16bf;
typedef __attribute__((ext_vector_type(8)))  float  v8f;
typedef __attribute__((ext_vector_type(2)))  __bf16 bf16x2;
typedef __attribute__((ext_vector_type(4)))  unsigned int u32x4;
typedef __attribute__((ext_vector_type(4)))  int i32x4;
typedef __attribute__((ext_vector_type(8)))  int i32x8;

#define B_  2
#define S_  2048
#define D_  1024
#define H_  16
#define HD_ 64
#define NEG_ (-1.0e9f)

#if defined(__clang_major__) && (__clang_major__ >= 23)
#define TDM_LOAD(g0, g1) __builtin_amdgcn_tensor_load_to_lds((g0), (g1), (i32x4)0, (i32x4)0, (i32x8)0, 0)
#else
#define TDM_LOAD(g0, g1) __builtin_amdgcn_tensor_load_to_lds((g0), (g1), (i32x4)0, (i32x4)0, 0)
#endif

__device__ __forceinline__ int lane_id() { return (int)(threadIdx.x & 31); }

// ---- TDM: async-copy a 2D row-major bf16 tile (tile_x cols, tile_rows rows)
// from global (row stride row_stride elems) into LDS at lds_dst. Wave-level op.
__device__ __forceinline__ void tdm_load_tile_2d(const __bf16* gsrc, const __bf16* lds_dst,
                                                 unsigned tile_x, unsigned tile_rows,
                                                 unsigned row_stride,
                                                 unsigned tensor_d0, unsigned tensor_d1) {
  const unsigned long long ga = (unsigned long long)(size_t)gsrc;
  const unsigned lo = (unsigned)(size_t)(const void*)lds_dst;   // low 32 bits = LDS byte offset
  u32x4 g0;
  g0.x = 1u;                                                    // count=1 (valid), no gather
  g0.y = lo;                                                    // lds_addr
  g0.z = (unsigned)ga;                                          // global_addr[31:0]
  g0.w = (unsigned)((ga >> 32) & 0x01FFFFFFu) | (2u << 30);     // global_addr[56:32] | type=2
  i32x8 g1;
  g1[0] = (int)(1u << 16);                                      // data_size = 2 bytes
  g1[1] = (int)((tensor_d0 & 0xFFFFu) << 16);                   // tensor_dim0[15:0] @ [31:16]
  g1[2] = (int)((tensor_d0 >> 16) | ((tensor_d1 & 0xFFFFu) << 16));
  g1[3] = (int)((tensor_d1 >> 16) | (tile_x << 16));            // tile_dim0 @ [127:112]
  g1[4] = (int)(tile_rows & 0xFFFFu);                           // tile_dim1 (tile_dim2 = 0)
  g1[5] = (int)row_stride;                                      // tensor_dim0_stride[31:0]
  g1[6] = 0;                                                    // stride[47:32] | dim1_stride lo
  g1[7] = 0;
  TDM_LOAD(g0, g1);
}

// Gather a 16x32 bf16 A / B^T fragment from a row-major LDS tile.
// CDNA5 16-bit layout: lanes 0-15 row=l, K {0..7,16..23}; lanes 16-31 row=l-16,
// K {8..15,24..31}; 2 bf16 per VGPR.
__device__ __forceinline__ v16bf load_frag(const __bf16* base, int row0, int stride, int kofs) {
  const int l  = lane_id();
  const int r  = row0 + (l & 15);
  const int kb = ((l >> 4) & 1) * 8;
  const __bf16* p = base + r * stride + kofs + kb;
  v16bf f;
#pragma unroll
  for (int i = 0; i < 8; ++i) {
    const int k = ((i & 4) << 2) + 2 * (i & 3);  // 0,2,4,6,16,18,20,22
    bf16x2 d = *(const bf16x2*)(p + k);
    f[2 * i]     = d.x;
    f[2 * i + 1] = d.y;
  }
  return f;
}

__device__ __forceinline__ v8f wmma_bf16(v16bf a, v16bf b, v8f c) {
  return __builtin_amdgcn_wmma_f32_16x16x32_bf16(false, a, false, b, (short)0, c, false, false);
}

// ---------------------------------------------------------------- converts
__global__ void cvt_f32_bf16(const float* __restrict__ in, __bf16* __restrict__ out, int n) {
  int i = blockIdx.x * blockDim.x + threadIdx.x;
  const int stride = gridDim.x * blockDim.x;
  for (; i < n; i += stride) out[i] = (__bf16)in[i];
}

// out[n*K + k] = (bf16) in[k*N + n]  -- 32x32 LDS tile transpose
__global__ __launch_bounds__(256)
void cvt_transpose_bf16(const float* __restrict__ in, __bf16* __restrict__ out, int K, int N) {
  __shared__ __bf16 t[32][33];
  const int n0 = blockIdx.x * 32, k0 = blockIdx.y * 32;
  const int tx = threadIdx.x & 31, ty = threadIdx.x >> 5;   // 8 rows of 32
#pragma unroll
  for (int i = ty; i < 32; i += 8) t[i][tx] = (__bf16)in[(size_t)(k0 + i) * N + n0 + tx];
  __syncthreads();
#pragma unroll
  for (int i = ty; i < 32; i += 8) out[(size_t)(n0 + i) * K + k0 + tx] = t[tx][i];
}

// ---------------------------------------------------------------- GEMM
// C[M,N] = A[M,K] * W[K,N] + bias, with W supplied pre-transposed Wt[N][K].
// Block tile 64x64, 4 waves of 32x32, TDM double-buffered LDS tiles.
// mode 0: bf16 out, [B,H,S,HD] remap (Q,K).  mode 2: bf16 out, [B,H,HD,S] (V).
// mode 1: fp32 out row-major (final projection).
__global__ __launch_bounds__(128)
void gemm_bf16_wmma(const __bf16* __restrict__ A, const __bf16* __restrict__ Wt,
                    const float* __restrict__ bias,
                    __bf16* __restrict__ outH, float* __restrict__ outF,
                    int mode, int M, int N, int K) {
  __shared__ __bf16 lA[2][64 * 32];   // [m][k]
  __shared__ __bf16 lB[2][64 * 32];   // [n][k]
  const int tid = threadIdx.x;
  const int wid = tid >> 5;
  const int wm = wid >> 1, wn = wid & 1;
  const int l  = tid & 31;
  const int hi = (l >> 4) & 1;
  const int r0 = blockIdx.y * 64;
  const int c0 = blockIdx.x * 64;

  __builtin_prefetch(bias + c0, 0, 1);

  if (wid == 0) {
    tdm_load_tile_2d(A  + (size_t)r0 * K, lA[0], 32, 64, K, K, M);
    tdm_load_tile_2d(Wt + (size_t)c0 * K, lB[0], 32, 64, K, K, N);
  }

  v8f acc[2][2] = {};
  const int nk = K / 32;
  for (int kt = 0; kt < nk; ++kt) {
    __builtin_amdgcn_s_wait_tensorcnt(0);
    __syncthreads();
    if (wid == 0 && kt + 1 < nk) {
      const int k1 = (kt + 1) * 32;
      tdm_load_tile_2d(A  + (size_t)r0 * K + k1, lA[(kt + 1) & 1], 32, 64, K, K, M);
      tdm_load_tile_2d(Wt + (size_t)c0 * K + k1, lB[(kt + 1) & 1], 32, 64, K, K, N);
    }
    const __bf16* la = lA[kt & 1];
    const __bf16* lb = lB[kt & 1];

    v16bf af[2], bfr[2];
#pragma unroll
    for (int i = 0; i < 2; ++i) af[i]  = load_frag(la, wm * 32 + i * 16, 32, 0);
#pragma unroll
    for (int j = 0; j < 2; ++j) bfr[j] = load_frag(lb, wn * 32 + j * 16, 32, 0);
#pragma unroll
    for (int i = 0; i < 2; ++i)
#pragma unroll
      for (int j = 0; j < 2; ++j)
        acc[i][j] = wmma_bf16(af[i], bfr[j], acc[i][j]);
  }

#pragma unroll
  for (int i = 0; i < 2; ++i) {
#pragma unroll
    for (int j = 0; j < 2; ++j) {
      const int c  = c0 + wn * 32 + j * 16 + (l & 15);
      const float bv = bias[c];
      const int h  = c >> 6;
      const int hd = c & 63;
#pragma unroll
      for (int v = 0; v < 8; ++v) {
        const int r = r0 + wm * 32 + i * 16 + v + 8 * hi;
        const float val = acc[i][j][v] + bv;
        if (mode == 0) {            // [B,H,S,HD]
          const int b = r >> 11, s = r & 2047;
          outH[(((size_t)(b * H_ + h)) * S_ + s) * HD_ + hd] = (__bf16)val;
        } else if (mode == 2) {     // [B,H,HD,S]
          const int b = r >> 11, s = r & 2047;
          outH[(((size_t)(b * H_ + h)) * HD_ + hd) * S_ + s] = (__bf16)val;
        } else {
          outF[(size_t)r * N + c] = val;
        }
      }
    }
  }
}

// ---------------------------------------------------------------- flash attention
// Block = (b, h, 64-row q tile), 128 threads. Wave w owns rows w*16..w*16+15
// across the full 64-wide key/hd dimension => softmax stats live in registers
// (shfl reductions within 16-lane groups). K and V^T tiles TDM double-buffered.
__global__ __launch_bounds__(128)
void flash_attn(const __bf16* __restrict__ Qg, const __bf16* __restrict__ Kg,
                const __bf16* __restrict__ Vt, const float* __restrict__ maskG,
                __bf16* __restrict__ Og) {
  __shared__ __bf16 lQ[64 * 64];        // [q][hd]
  __shared__ __bf16 lK[2][64 * 64];     // [key][hd]  (B^T for Q*K^T)
  __shared__ __bf16 lVT[2][64 * 64];    // [hd][key]  (B^T for P*V)
  __shared__ __bf16 lP[64 * 64];        // [q][key]
  __shared__ float  lMask[64];

  const int tid = threadIdx.x;
  const int wid = tid >> 5;
  const int l   = tid & 31;
  const int hi  = (l >> 4) & 1;

  const int qt = blockIdx.x;
  const int h  = blockIdx.y;
  const int b  = blockIdx.z;
  const int q0 = qt * 64;
  const size_t head  = ((size_t)b * H_ + h) * (size_t)S_ * HD_;   // Q,K layout
  const size_t headT = ((size_t)b * H_ + h) * (size_t)HD_ * S_;   // V^T layout

  __builtin_prefetch(maskG + b * S_, 0, 1);

  if (wid == 0) {
    tdm_load_tile_2d(Qg + head + (size_t)q0 * HD_, lQ,     64, 64, HD_, HD_, S_);
    tdm_load_tile_2d(Kg + head,                    lK[0],  64, 64, HD_, HD_, S_);
    tdm_load_tile_2d(Vt + headT,                   lVT[0], 64, 64, S_,  S_,  HD_);
  }

  v8f  o[4] = {};              // 16 rows x 64 hd per wave
  float mrow[8], lrow[8];      // per (v, half-wave) row stats
#pragma unroll
  for (int v = 0; v < 8; ++v) { mrow[v] = -INFINITY; lrow[v] = 0.0f; }

  const float sm_scale = 0.125f;   // 1/sqrt(HD)

  for (int kt = 0; kt <= qt; ++kt) {
    const int k0 = kt * 64;
    __builtin_amdgcn_s_wait_tensorcnt(0);
    __syncthreads();
    if (tid < 64) lMask[tid] = maskG[b * S_ + k0 + tid];
    if (wid == 0 && kt < qt) {
      const int k1 = k0 + 64;
      tdm_load_tile_2d(Kg + head + (size_t)k1 * HD_, lK[(kt + 1) & 1],  64, 64, HD_, HD_, S_);
      tdm_load_tile_2d(Vt + headT + k1,              lVT[(kt + 1) & 1], 64, 64, S_,  S_,  HD_);
    }
    __syncthreads();
    const __bf16* lk = lK[kt & 1];
    const __bf16* lv = lVT[kt & 1];

    // ---- S = (Q K^T) * scale + masks ; wave rows wid*16..+15, key tiles j=0..3
    v8f sacc[4] = {};
#pragma unroll
    for (int kk = 0; kk < 2; ++kk) {
      v16bf aq = load_frag(lQ, wid * 16, 64, kk * 32);
#pragma unroll
      for (int j = 0; j < 4; ++j) {
        v16bf bk = load_frag(lk, j * 16, 64, kk * 32);
        sacc[j] = wmma_bf16(aq, bk, sacc[j]);
      }
    }
#pragma unroll
    for (int j = 0; j < 4; ++j) {
      const int n  = j * 16 + (l & 15);
      const int kg = k0 + n;
      const float mpad = (1.0f - lMask[n]) * NEG_;
#pragma unroll
      for (int v = 0; v < 8; ++v) {
        const int qg = q0 + wid * 16 + v + 8 * hi;
        float val = sacc[j][v] * sm_scale;
        val = (kg <= qg) ? val : NEG_;
        sacc[j][v] = val + mpad;
      }
    }

    // ---- online softmax: row = wid*16 + v + 8*hi, reduce over 16-lane groups
    float mnew[8], sc[8];
#pragma unroll
    for (int v = 0; v < 8; ++v) {
      float rm = fmaxf(fmaxf(sacc[0][v], sacc[1][v]), fmaxf(sacc[2][v], sacc[3][v]));
#pragma unroll
      for (int off = 8; off >= 1; off >>= 1) rm = fmaxf(rm, __shfl_xor(rm, off, 16));
      mnew[v] = fmaxf(mrow[v], rm);
      sc[v]   = __expf(mrow[v] - mnew[v]);
      mrow[v] = mnew[v];
    }
#pragma unroll
    for (int v = 0; v < 8; ++v) {
      float rs = 0.0f;
#pragma unroll
      for (int j = 0; j < 4; ++j) {
        const float p = __expf(sacc[j][v] - mnew[v]);
        sacc[j][v] = p;
        rs += p;
      }
#pragma unroll
      for (int off = 8; off >= 1; off >>= 1) rs += __shfl_xor(rs, off, 16);
      lrow[v] = lrow[v] * sc[v] + rs;
    }

    // ---- write P (bf16) and rescale O
    const int rbase = wid * 16 + 8 * hi;
#pragma unroll
    for (int j = 0; j < 4; ++j) {
      const int n = j * 16 + (l & 15);
#pragma unroll
      for (int v = 0; v < 8; ++v) lP[(rbase + v) * 64 + n] = (__bf16)sacc[j][v];
    }
#pragma unroll
    for (int j = 0; j < 4; ++j)
#pragma unroll
      for (int v = 0; v < 8; ++v) o[j][v] *= sc[v];
    __syncthreads();

    // ---- O += P V   (A = lP rows, B^T = lVT rows over hd)
#pragma unroll
    for (int kk = 0; kk < 2; ++kk) {
      v16bf ap = load_frag(lP, wid * 16, 64, kk * 32);
#pragma unroll
      for (int j = 0; j < 4; ++j) {
        v16bf bv = load_frag(lv, j * 16, 64, kk * 32);
        o[j] = wmma_bf16(ap, bv, o[j]);
      }
    }
  }

  // ---- normalize, store to [B,S,D] bf16
#pragma unroll
  for (int j = 0; j < 4; ++j) {
    const int n = j * 16 + (l & 15);
#pragma unroll
    for (int v = 0; v < 8; ++v) {
      const int r = wid * 16 + v + 8 * hi;
      const float inv = 1.0f / lrow[v];
      Og[((size_t)b * S_ + q0 + r) * D_ + h * HD_ + n] = (__bf16)(o[j][v] * inv);
    }
  }
}

// ---------------------------------------------------------------- launch
extern "C" void kernel_launch(void* const* d_in, const int* in_sizes, int n_in,
                              void* d_out, int out_size, void* d_ws, size_t ws_size,
                              hipStream_t stream) {
  (void)in_sizes; (void)n_in; (void)out_size; (void)ws_size;
  const float* x    = (const float*)d_in[0];
  const float* mask = (const float*)d_in[1];
  const float* Wq   = (const float*)d_in[2];
  const float* bq   = (const float*)d_in[3];
  const float* Wk   = (const float*)d_in[4];
  const float* bk   = (const float*)d_in[5];
  const float* Wv   = (const float*)d_in[6];
  const float* bv   = (const float*)d_in[7];
  const float* Wo   = (const float*)d_in[8];
  const float* bo   = (const float*)d_in[9];
  float* out = (float*)d_out;

  char* ws = (char*)d_ws;
  __bf16* xb  = (__bf16*)(ws);                          //  8 MB  x bf16 [4096,1024]
  __bf16* wqT = (__bf16*)(ws + ((size_t)8  << 20));     //  2 MB  Wq^T bf16 [N][K]
  __bf16* wkT = (__bf16*)(ws + ((size_t)10 << 20));
  __bf16* wvT = (__bf16*)(ws + ((size_t)12 << 20));
  __bf16* woT = (__bf16*)(ws + ((size_t)14 << 20));
  __bf16* Qb  = (__bf16*)(ws + ((size_t)16 << 20));     //  8 MB  [B,H,S,HD]
  __bf16* Kb  = (__bf16*)(ws + ((size_t)24 << 20));     //  8 MB  [B,H,S,HD]
  __bf16* Vtb = (__bf16*)(ws + ((size_t)32 << 20));     //  8 MB  [B,H,HD,S]
  __bf16* Ab  = (__bf16*)(ws + ((size_t)40 << 20));     //  8 MB  attn out [B,S,D]

  const int NX = B_ * S_ * D_;
  hipLaunchKernelGGL(cvt_f32_bf16, dim3(2048), dim3(256), 0, stream, x, xb, NX);
  const dim3 tg(D_ / 32, D_ / 32);                      // (32, 32)
  hipLaunchKernelGGL(cvt_transpose_bf16, tg, dim3(256), 0, stream, Wq, wqT, D_, D_);
  hipLaunchKernelGGL(cvt_transpose_bf16, tg, dim3(256), 0, stream, Wk, wkT, D_, D_);
  hipLaunchKernelGGL(cvt_transpose_bf16, tg, dim3(256), 0, stream, Wv, wvT, D_, D_);
  hipLaunchKernelGGL(cvt_transpose_bf16, tg, dim3(256), 0, stream, Wo, woT, D_, D_);

  const dim3 gg(D_ / 64, (B_ * S_) / 64);               // (16, 64)
  const dim3 gb(128);
  hipLaunchKernelGGL(gemm_bf16_wmma, gg, gb, 0, stream, xb, wqT, bq, Qb,  (float*)nullptr, 0, B_ * S_, D_, D_);
  hipLaunchKernelGGL(gemm_bf16_wmma, gg, gb, 0, stream, xb, wkT, bk, Kb,  (float*)nullptr, 0, B_ * S_, D_, D_);
  hipLaunchKernelGGL(gemm_bf16_wmma, gg, gb, 0, stream, xb, wvT, bv, Vtb, (float*)nullptr, 2, B_ * S_, D_, D_);

  const dim3 fg(S_ / 64, H_, B_);                       // (32, 16, 2)
  hipLaunchKernelGGL(flash_attn, fg, gb, 0, stream, Qb, Kb, Vtb, mask, Ab);

  hipLaunchKernelGGL(gemm_bf16_wmma, gg, gb, 0, stream, Ab, woT, bo, (__bf16*)nullptr, out, 1, B_ * S_, D_, D_);
}
